// Attention_4243427689031
// MI455X (gfx1250) — matrix operand order repaired and checked by reading, not compiled, not run on hardware
//
#include <hip/hip_runtime.h>
#include <hip/hip_bf16.h>

typedef __attribute__((ext_vector_type(16))) __bf16 v16bf;
typedef __attribute__((ext_vector_type(8)))  __bf16 v8bf;
typedef __attribute__((ext_vector_type(8)))  float  v8f;
typedef __attribute__((ext_vector_type(4)))  int    v4i;

#define AS1 __attribute__((address_space(1)))
#define AS3 __attribute__((address_space(3)))

#define NH      12
#define HD      64
#define NTOK    196
#define BATCH   128
#define MROWS   (BATCH * NTOK)      // 25088
#define CDIM    768
#define NPADR   208                 // 13 * 16  (row padding for S)
#define MPADC   224                 // 7 * 32   (col padding for S, K-dim of PV)

// ---------------------------------------------------------------------------
// WMMA helpers (CDNA5: D = A(16x32 bf16) * B(32x16 bf16) + C(16x16 f32))
// ---------------------------------------------------------------------------
__device__ __forceinline__ v8f wmma_bf16(v16bf a, v16bf b, v8f c) {
  return __builtin_amdgcn_wmma_f32_16x16x32_bf16(
      /*neg_a=*/false, a, /*neg_b=*/false, b,
      /*c_mod=*/(short)0, c, /*reuse_a=*/false, /*reuse_b=*/false);
}

// A fragment: lane 0-15 -> row M=lane, K {0..7,16..23}; lane 16-31 -> K {8..15,24..31}
__device__ __forceinline__ v16bf lds_a_frag(const __bf16* base, int row0, int k0,
                                            int ld, int lane) {
  const int kb = (lane < 16) ? 0 : 8;
  const __bf16* p = base + (row0 + (lane & 15)) * ld + k0 + kb;
  v8bf lo = *(const v8bf*)p;
  v8bf hi = *(const v8bf*)(p + 16);
  return __builtin_shufflevector(lo, hi, 0, 1, 2, 3, 4, 5, 6, 7,
                                 8, 9, 10, 11, 12, 13, 14, 15);
}

// Same A fragment but converting f32 (softmax probabilities) -> bf16 on the fly
__device__ __forceinline__ v16bf lds_a_frag_f32(const float* base, int row0, int k0,
                                                int ld, int lane) {
  const int kb = (lane < 16) ? 0 : 8;
  const float* p = base + (row0 + (lane & 15)) * ld + k0 + kb;
  v16bf r;
#pragma unroll
  for (int j = 0; j < 8; ++j) {
    r[j]     = (__bf16)p[j];
    r[j + 8] = (__bf16)p[j + 16];
  }
  return r;
}

// B fragment: lane 0-15 -> col N=lane, K {0..7,16..23}; lane 16-31 -> K {8..15,24..31}.
__device__ __forceinline__ v16bf lds_b_frag(const __bf16* base, int col0, int k0,
                                            int ld, int lane) {
  const int kb = (lane < 16) ? 0 : 8;
  const __bf16* p = base + (col0 + (lane & 15)) * ld + k0 + kb;
  v8bf lo = *(const v8bf*)p;
  v8bf hi = *(const v8bf*)(p + 16);
  return __builtin_shufflevector(lo, hi, 0, 1, 2, 3, 4, 5, 6, 7,
                                 8, 9, 10, 11, 12, 13, 14, 15);
}

template <typename AT>
__device__ __forceinline__ void ld8(const AT* src, float* d) {
#pragma unroll
  for (int j = 0; j < 8; ++j) d[j] = (float)src[j];
}
__device__ __forceinline__ void st8(__bf16* dst, const float* s) {
  v8bf t;
#pragma unroll
  for (int j = 0; j < 8; ++j) t[j] = (__bf16)s[j];
  *(v8bf*)dst = t;
}

// ---------------------------------------------------------------------------
// GEMM: C[M,N] = A[M,K] * W[N,K]^T
// Block tile 128x128, BK=32, 8 waves each 32x64 (8 WMMA / K-step).
// Register + LDS double buffering: one barrier per K-step.
// EPI == 0 : QKV epilogue -> bf16 [3][B][h][N][d] buffer with bias/scale
// EPI == 1 : proj epilogue -> f32 out + bias
// ---------------------------------------------------------------------------
template <int EPI, typename AT>
__global__ __launch_bounds__(256) void gemm_wmma(
    const AT* __restrict__ A, const float* __restrict__ W,
    int M, int N, int K,
    __bf16* __restrict__ qkv_out, const float* __restrict__ q_bias,
    const float* __restrict__ v_bias,
    float* __restrict__ f32_out, const float* __restrict__ bias) {
  (void)M;
  __shared__ __bf16 As[2][128 * 32];
  __shared__ __bf16 Bs[2][128 * 32];

  const int tid  = threadIdx.x;
  const int lane = tid & 31;
  const int w    = tid >> 5;
  const int wm   = (w & 3) * 32;   // wave row offset in block tile
  const int wn   = (w >> 2) * 64;  // wave col offset
  const long bm  = (long)blockIdx.y * 128;
  const long bn  = (long)blockIdx.x * 128;

  // staging chunks: 512 8-elem chunks per 128x32 tile -> 2 per thread
  const int r0 = tid >> 2;          // 0..63
  const int r1 = r0 + 64;           // 64..127
  const int kc = (tid & 3) * 8;     // 0,8,16,24

  float aR0[8], aR1[8], bR0[8], bR1[8];
  v8f acc[2][4] = {{{}, {}, {}, {}}, {{}, {}, {}, {}}};

  auto load_tiles = [&](int k0) {
    ld8(A + (bm + r0) * (long)K + k0 + kc, aR0);
    ld8(A + (bm + r1) * (long)K + k0 + kc, aR1);
    ld8(W + (bn + r0) * (long)K + k0 + kc, bR0);
    ld8(W + (bn + r1) * (long)K + k0 + kc, bR1);
  };
  auto store_tiles = [&](int buf) {
    st8(&As[buf][r0 * 32 + kc], aR0);
    st8(&As[buf][r1 * 32 + kc], aR1);
    st8(&Bs[buf][r0 * 32 + kc], bR0);
    st8(&Bs[buf][r1 * 32 + kc], bR1);
  };
  auto compute = [&](int buf) {
    const __bf16* as = As[buf];
    const __bf16* bs = Bs[buf];
    v16bf a0 = lds_a_frag(as, wm, 0, 32, lane);
    v16bf a1 = lds_a_frag(as, wm + 16, 0, 32, lane);
#pragma unroll
    for (int j = 0; j < 4; ++j) {
      v16bf bf = lds_b_frag(bs, wn + j * 16, 0, 32, lane);
      acc[0][j] = wmma_bf16(a0, bf, acc[0][j]);
      acc[1][j] = wmma_bf16(a1, bf, acc[1][j]);
    }
  };

  load_tiles(0);
  store_tiles(0);
  __syncthreads();
  int cur = 0;
  for (int k0 = 32; k0 < K; k0 += 32) {
    load_tiles(k0);        // global loads for tile k0 in flight ...
    compute(cur);          // ... while WMMAs run on tile k0-32
    store_tiles(cur ^ 1);
    __syncthreads();
    cur ^= 1;
  }
  compute(cur);

  // ---------------- epilogue ----------------
  const int rofs = (lane >= 16) ? 8 : 0;
  const int cofs = lane & 15;
  if constexpr (EPI == 0) {
    int tt[4], hh[4], dd[4];
    float badd[4], mul[4];
#pragma unroll
    for (int sn = 0; sn < 4; ++sn) {
      const int gn = (int)(bn + wn + sn * 16 + cofs);
      const int t = gn / CDIM, cc = gn % CDIM;
      tt[sn] = t; hh[sn] = cc >> 6; dd[sn] = cc & 63;
      badd[sn] = (t == 0) ? q_bias[cc] : (t == 2 ? v_bias[cc] : 0.f);
      mul[sn]  = (t == 0) ? 0.125f : 1.f;  // head_dim^-0.5
    }
#pragma unroll
    for (int sm = 0; sm < 2; ++sm) {
#pragma unroll
      for (int r = 0; r < 8; ++r) {
        const long gm = bm + wm + sm * 16 + r + rofs;
        const int b = (int)(gm / NTOK), n = (int)(gm % NTOK);
#pragma unroll
        for (int sn = 0; sn < 4; ++sn) {
          const float val = (acc[sm][sn][r] + badd[sn]) * mul[sn];
          qkv_out[((((long)tt[sn] * BATCH + b) * NH + hh[sn]) * NTOK + n) * HD +
                  dd[sn]] = (__bf16)val;
        }
      }
    }
  } else {
    long gncol[4];
    float bb[4];
#pragma unroll
    for (int sn = 0; sn < 4; ++sn) {
      gncol[sn] = bn + wn + sn * 16 + cofs;
      bb[sn] = bias[gncol[sn]];
    }
#pragma unroll
    for (int sm = 0; sm < 2; ++sm) {
#pragma unroll
      for (int r = 0; r < 8; ++r) {
        const long gm = bm + wm + sm * 16 + r + rofs;
#pragma unroll
        for (int sn = 0; sn < 4; ++sn)
          f32_out[gm * N + gncol[sn]] = acc[sm][sn][r] + bb[sn];
      }
    }
  }
}

// ---------------------------------------------------------------------------
// Attention: one workgroup (256 thr, 8 waves) per (batch, head).
// Whole 196x196 score matrix lives in LDS; softmax + rel-pos bias fused.
// ---------------------------------------------------------------------------
__global__ __launch_bounds__(256) void attn_kernel(
    const __bf16* __restrict__ qkv,       // [3][B][NH][NTOK][HD] bf16
    const float* __restrict__ rel_pos_h,  // [27][64]
    const float* __restrict__ rel_pos_w,  // [27][64]
    __bf16* __restrict__ attn_out) {      // [B*NTOK][CDIM] bf16
  extern __shared__ char smem[];
  float*  S    = (float*)smem;                                   // [208][224]
  __bf16* Vt   = (__bf16*)(smem + NPADR * MPADC * 4);            // [64][224]
  __bf16* Qb   = (__bf16*)((char*)Vt + HD * MPADC * 2);          // [208][64]
  __bf16* Kb   = (__bf16*)((char*)Qb + NPADR * HD * 2);          // [208][64]
  float*  relh = (float*)((char*)Kb + NPADR * HD * 2);           // [196][14]
  float*  relw = relh + NTOK * 14;                               // [196][14]

  const int bh = blockIdx.x;
  const int b  = bh / NH, h = bh % NH;
  const long TSZ = (long)BATCH * NH * NTOK * HD;
  const __bf16* qg = qkv + ((long)b * NH + h) * NTOK * HD;
  const __bf16* kg = qg + TSZ;
  const __bf16* vg = qg + 2 * TSZ;

  const int tid  = threadIdx.x;
  const int lane = tid & 31;
  const int w    = tid >> 5;

  // Phase 1: stage Q and K to LDS — async DMA path (ASYNCcnt) when available
  {
    const uint4* qs = (const uint4*)qg;
    const uint4* ks = (const uint4*)kg;
    uint4* qd = (uint4*)Qb;
    uint4* kd = (uint4*)Kb;
#if __has_builtin(__builtin_amdgcn_global_load_async_to_lds_b128)
    for (int i = tid; i < NTOK * HD / 8; i += 256) {
      __builtin_amdgcn_global_load_async_to_lds_b128(
          (AS1 v4i*)(qs + i), (AS3 v4i*)(qd + i), 0, 0);
      __builtin_amdgcn_global_load_async_to_lds_b128(
          (AS1 v4i*)(ks + i), (AS3 v4i*)(kd + i), 0, 0);
    }
#if __has_builtin(__builtin_amdgcn_s_wait_asynccnt)
    __builtin_amdgcn_s_wait_asynccnt(0);
#else
    asm volatile("s_wait_asynccnt 0" ::: "memory");
#endif
#else
    for (int i = tid; i < NTOK * HD / 8; i += 256) {
      qd[i] = qs[i];
      kd[i] = ks[i];
    }
#endif
  }
  __syncthreads();

  // Phase 2a: rel_h / rel_w tables  rel[n][kk] = sum_c q[n][c]*rel_pos[dist][c]
  for (int idx = tid; idx < NTOK * 28; idx += 256) {
    const int n   = idx / 28;
    const int rem = idx % 28;
    const int kk  = rem % 14;
    const bool iw = rem >= 14;
    const int qi = n / 14, qj = n % 14;
    const int dpos = (iw ? qj : qi) - kk + 13;
    const float*  rp   = (iw ? rel_pos_w : rel_pos_h) + dpos * HD;
    const __bf16* qrow = Qb + n * HD;
    float s = 0.f;
#pragma unroll 8
    for (int c = 0; c < HD; ++c) s += (float)qrow[c] * rp[c];
    (iw ? relw : relh)[n * 14 + kk] = s;
  }

  // Phase 2b: S = Q * K^T; wave owns a row-tile, A-fragments hoisted over tn
  for (int tm = w; tm < 13; tm += 8) {
    v16bf a0 = lds_a_frag(Qb, tm * 16, 0, HD, lane);
    v16bf a1 = lds_a_frag(Qb, tm * 16, 32, HD, lane);
    const int rofs = (lane >= 16) ? 8 : 0;
    for (int tn = 0; tn < 13; ++tn) {
      v8f c = {};
      c = wmma_bf16(a0, lds_b_frag(Kb, tn * 16, 0, HD, lane), c);
      c = wmma_bf16(a1, lds_b_frag(Kb, tn * 16, 32, HD, lane), c);
      const int col = tn * 16 + (lane & 15);
#pragma unroll
      for (int r = 0; r < 8; ++r)
        S[(tm * 16 + r + rofs) * MPADC + col] = c[r];
    }
  }
  __syncthreads();

  // Phase 3a: stage V transposed (zero-padded K-dim 196..223)
  for (int idx = tid; idx < HD * MPADC; idx += 256) {
    const int d = idx / MPADC, m = idx % MPADC;
    Vt[idx] = (m < NTOK) ? vg[(long)m * HD + d] : (__bf16)0.f;
  }
  // Phase 3b: fused rel-bias + softmax, wave per row, shfl_xor reductions
  for (int row = w; row < NPADR; row += 8) {
    float* r = S + row * MPADC;
    if (row < NTOK) {
      const float* rh = relh + row * 14;
      const float* rw = relw + row * 14;
      float sv[7];
      float mx = -1e30f;
#pragma unroll
      for (int i = 0; i < 7; ++i) {
        const int m = lane + 32 * i;
        const float s = (m < NTOK) ? (r[m] + rh[m / 14] + rw[m % 14]) : -1e30f;
        sv[i] = s;
        mx = fmaxf(mx, s);
      }
#pragma unroll
      for (int off = 16; off; off >>= 1) mx = fmaxf(mx, __shfl_xor(mx, off, 32));
      float sum = 0.f;
#pragma unroll
      for (int i = 0; i < 7; ++i) {
        const int m = lane + 32 * i;
        const float e = (m < NTOK) ? __expf(sv[i] - mx) : 0.f;
        sv[i] = e;
        sum += e;
      }
#pragma unroll
      for (int off = 16; off; off >>= 1) sum += __shfl_xor(sum, off, 32);
      const float inv = 1.f / sum;
#pragma unroll
      for (int i = 0; i < 7; ++i) r[lane + 32 * i] = sv[i] * inv;
    } else {
#pragma unroll
      for (int i = 0; i < 7; ++i) r[lane + 32 * i] = 0.f;
    }
  }
  __syncthreads();

  // Phase 4: O = P * V; wave owns a row-tile, P-fragment reused over 4 d-tiles
  for (int tm = w; tm < 13; tm += 8) {
    v8f acc4[4] = {{}, {}, {}, {}};
#pragma unroll
    for (int k0 = 0; k0 < MPADC; k0 += 32) {
      v16bf a = lds_a_frag_f32(S, tm * 16, k0, MPADC, lane);
#pragma unroll
      for (int td = 0; td < 4; ++td)
        acc4[td] = wmma_bf16(a, lds_b_frag(Vt, td * 16, k0, MPADC, lane),
                             acc4[td]);
    }
    const int rofs = (lane >= 16) ? 8 : 0;
#pragma unroll
    for (int td = 0; td < 4; ++td) {
      const int d = td * 16 + (lane & 15);
#pragma unroll
      for (int r = 0; r < 8; ++r) {
        const int n = tm * 16 + r + rofs;
        if (n < NTOK)
          attn_out[((long)b * NTOK + n) * CDIM + h * HD + d] = (__bf16)acc4[td][r];
      }
    }
  }
}

// ---------------------------------------------------------------------------
// Launch
// ---------------------------------------------------------------------------
extern "C" void kernel_launch(void* const* d_in, const int* in_sizes, int n_in,
                              void* d_out, int out_size, void* d_ws,
                              size_t ws_size, hipStream_t stream) {
  (void)in_sizes; (void)n_in; (void)out_size; (void)ws_size;
  const float* x      = (const float*)d_in[0];
  const float* qkv_w  = (const float*)d_in[1];
  const float* q_bias = (const float*)d_in[2];
  const float* v_bias = (const float*)d_in[3];
  const float* rel_h  = (const float*)d_in[4];
  const float* rel_w  = (const float*)d_in[5];
  const float* proj_w = (const float*)d_in[6];
  const float* proj_b = (const float*)d_in[7];

  // Workspace: qkv bf16 (115.6 MB) + attention output bf16 (38.5 MB)
  __bf16* qkv_buf  = (__bf16*)d_ws;
  __bf16* attn_buf = qkv_buf + (size_t)3 * BATCH * NH * NTOK * HD;

  dim3 blk(256);

  // QKV projection: [25088,768] x [2304,768]^T
  gemm_wmma<0, float><<<dim3(3 * CDIM / 128, MROWS / 128), blk, 0, stream>>>(
      x, qkv_w, MROWS, 3 * CDIM, CDIM,
      qkv_buf, q_bias, v_bias, nullptr, nullptr);

  // Attention: one WGP workgroup per (b, h), ~283 KB LDS of the 320 KB/WGP
  constexpr size_t ATTN_SMEM = (size_t)NPADR * MPADC * 4 + HD * MPADC * 2 +
                               2 * NPADR * HD * 2 + 2 * NTOK * 14 * 4;
  (void)hipFuncSetAttribute((const void*)attn_kernel,
                            hipFuncAttributeMaxDynamicSharedMemorySize,
                            (int)ATTN_SMEM);
  attn_kernel<<<dim3(BATCH * NH), blk, ATTN_SMEM, stream>>>(
      qkv_buf, rel_h, rel_w, attn_buf);

  // Output projection: [25088,768] x [768,768]^T + bias -> f32 out
  gemm_wmma<1, __bf16><<<dim3(CDIM / 128, MROWS / 128), blk, 0, stream>>>(
      attn_buf, proj_w, MROWS, CDIM, CDIM,
      nullptr, nullptr, nullptr, (float*)d_out, proj_b);
}
